// GAT_FP_41695542510280
// MI455X (gfx1250) — matrix operand order repaired and checked by reading, not compiled, not run on hardware
//
#include <hip/hip_runtime.h>
#include <stdint.h>

// ---------------------------------------------------------------------------
// Problem constants (match reference)
// ---------------------------------------------------------------------------
#define NNODES 10000
#define NEDGES 64000
#define INFEAT 1247
#define HEADS  4
#define DIM0   256
#define DIM1   8
#define OUTC   6

#define MPAD   10048           // 10000 rounded up to 64 (64-row wave tiles)
#define KP0    1248            // 1247 rounded up to 32
#define NB0    3104            // 1024 Wl0 | 1024 Wr0 | 1024 Wres0 | 32 W2 (97 col tiles of 32)
#define NCT0   (NB0 / 16)      // 194 sixteen-wide fragment column tiles
#define KP1    1024
#define NB1    96              // 32 Wl1 | 32 Wr1 | 32 Wres1
#define NCT1   (NB1 / 16)      // 6

typedef __attribute__((ext_vector_type(16))) __bf16 v16bf;
typedef __attribute__((ext_vector_type(8)))  float  v8f;

static inline int cdiv(long long a, long long b) { return (int)((a + b - 1) / b); }

// ---------------------------------------------------------------------------
// WMMA fragment helpers (ISA 7.12.2, wave32 16-bit layouts)
// ---------------------------------------------------------------------------
__device__ inline v16bf load_a_frag(const __bf16* A, int row0, int k0, int lda, int lane) {
  // A 16x32 (MxK): lane r (0-15) row M=r holds K=hi*16+{0..15}
  int r = lane & 15, hi = lane >> 4;
  const __bf16* base = A + (size_t)(row0 + r) * lda + k0 + hi * 16;
  v16bf a;
#pragma unroll
  for (int v = 0; v < 4; ++v) { a[2*v] = base[2*v]; a[2*v+1] = base[2*v+1]; }
#pragma unroll
  for (int v = 0; v < 4; ++v) { a[8+2*v] = base[8+2*v]; a[8+2*v+1] = base[8+2*v+1]; }
  return a;
}

// B panel is stored pre-swizzled: for (ktile, 16-col ctile), lane's 16 values
// (K = (lane>>4)*16 + e, col = ctile*16 + (lane&15)) are contiguous -> one
// aligned 32-byte load (2x global_load_b128), fully coalesced across lanes.
__device__ inline v16bf load_b_frag_swz(const __bf16* Bs, int kt, int ct, int nct, int lane) {
  const v16bf* p = (const v16bf*)(Bs + ((((size_t)kt * nct + ct) * 32 + lane) << 4));
  return *p;
}

__device__ inline void store_c_frag(float* C, const v8f& c, int row0, int col0, int ldc, int lane) {
  int n = lane & 15, hi = lane >> 4;
#pragma unroll
  for (int j = 0; j < 8; ++j)
    C[(size_t)(row0 + j + 8 * hi) * ldc + col0 + n] = c[j];
}

__device__ inline v8f wmma_bf16(const v16bf& a, const v16bf& b, const v8f& c) {
  return __builtin_amdgcn_wmma_f32_16x16x32_bf16(false, a, false, b, (short)0, c, false, false);
}

// ---------------------------------------------------------------------------
// WMMA GEMM: C[M,Nc] = A[M,K](bf16, row-major) x Bs[K,Nc](bf16, swizzled).
// Each wave owns one 64x32 output tile: 4 A frags x 2 B frags -> 8 WMMAs/iter.
// ---------------------------------------------------------------------------
__global__ void __launch_bounds__(256)
wmma_gemm_kernel(const __bf16* __restrict__ A, const __bf16* __restrict__ Bs,
                 float* __restrict__ C, int Mt /*M/64*/, int Nt /*Nc/32*/,
                 int K, int lda, int nct /*Nc/16*/, int ldc) {
  int wave = blockIdx.x * (blockDim.x >> 5) + (threadIdx.x >> 5);
  int lane = threadIdx.x & 31;
  int total = Mt * Nt;
  if (wave >= total) return;                 // wave-uniform branch (EXEC stays all-ones)
  int tm = wave / Nt, tn = wave % Nt;
  int row0 = tm * 64, col0 = tn * 32, ct0 = tn * 2;

  v8f acc[4][2] = {};
  int Kt = K >> 5;
  for (int kt = 0; kt < Kt; ++kt) {
    int k0 = kt << 5;
    if (kt + 1 < Kt) {                       // hint next K tile -> global_prefetch_b8
      __builtin_prefetch(A + (size_t)row0 * lda + k0 + 32, 0, 3);
      __builtin_prefetch(Bs + (((size_t)(kt + 1) * nct + ct0) * 32) * 16, 0, 3);
    }
    v16bf b0 = load_b_frag_swz(Bs, kt, ct0,     nct, lane);
    v16bf b1 = load_b_frag_swz(Bs, kt, ct0 + 1, nct, lane);
#pragma unroll
    for (int i = 0; i < 4; ++i) {
      v16bf a = load_a_frag(A, row0 + 16 * i, k0, lda, lane);
      acc[i][0] = wmma_bf16(a, b0, acc[i][0]);
      acc[i][1] = wmma_bf16(a, b1, acc[i][1]);
    }
  }
#pragma unroll
  for (int i = 0; i < 4; ++i) {
    store_c_frag(C, acc[i][0], row0 + 16 * i, col0,      ldc, lane);
    store_c_frag(C, acc[i][1], row0 + 16 * i, col0 + 16, ldc, lane);
  }
}

// ---------------------------------------------------------------------------
// Preprocessing
// ---------------------------------------------------------------------------
__global__ void col_mean_kernel(const float* __restrict__ f, float* __restrict__ mean) {
  int j = blockIdx.x * blockDim.x + threadIdx.x;
  if (j >= INFEAT) return;
  float s = 0.f;
  for (int i = 0; i < NNODES; ++i) s += f[(size_t)i * INFEAT + j];
  mean[j] = s / (float)NNODES;
}

// impute + blend + L1-normalize + modality masks, emit padded bf16 tile matrix
__global__ void preprocess_kernel(const float* __restrict__ f, const float* __restrict__ mean,
                                  const float* __restrict__ tm, const float* __restrict__ am,
                                  const float* __restrict__ vm, __bf16* __restrict__ Xbf) {
  int row = blockIdx.x;                      // [0, MPAD)
  __shared__ float red[256];
  float inv = 0.f;
  if (row < NNODES) {
    const float* fr = f + (size_t)row * INFEAT;
    float s = 0.f;
    for (int j = threadIdx.x; j < INFEAT; j += blockDim.x) {
      float x = fr[j];
      float h1 = (x == 0.0f) ? mean[j] : x;
      s += fabsf(0.5f * (x + h1));
    }
    red[threadIdx.x] = s;
    __syncthreads();
    for (int off = 128; off > 0; off >>= 1) {
      if (threadIdx.x < off) red[threadIdx.x] += red[threadIdx.x + off];
      __syncthreads();
    }
    inv = 1.0f / fmaxf(red[0], 1e-12f);
  }
  for (int j = threadIdx.x; j < KP0; j += blockDim.x) {
    float val = 0.f;
    if (row < NNODES && j < INFEAT) {
      float x = f[(size_t)row * INFEAT + j];
      float h1 = (x == 0.0f) ? mean[j] : x;
      val = 0.5f * (x + h1) * inv * (tm[j] + am[j] + vm[j]);
    }
    Xbf[(size_t)row * KP0 + j] = (__bf16)val;
  }
}

// pack [Wl0 | Wr0 | Wres0 | W2-reshaped] -> bf16 swizzled fragment panel
__global__ void pack_w0_kernel(const float* __restrict__ Wl0, const float* __restrict__ Wr0,
                               const float* __restrict__ Wres0, const float* __restrict__ W2,
                               __bf16* __restrict__ B0s) {
  size_t idx = (size_t)blockIdx.x * blockDim.x + threadIdx.x;
  if (idx >= (size_t)KP0 * NB0) return;
  int e    = (int)(idx & 15);
  int lane = (int)((idx >> 4) & 31);
  size_t tile = idx >> 9;
  int ct = (int)(tile % NCT0), kt = (int)(tile / NCT0);
  int k = kt * 32 + (lane >> 4) * 16 + e;
  int c = ct * 16 + (lane & 15);
  float v = 0.f;
  if (k < INFEAT) {
    if (c < 1024)      v = Wl0[(size_t)k * 1024 + c];
    else if (c < 2048) v = Wr0[(size_t)k * 1024 + (c - 1024)];
    else if (c < 3072) v = Wres0[(size_t)k * 1024 + (c - 2048)];
    else {
      int c2 = c - 3072, hh = c2 >> 3, o = c2 & 7;   // W2[h, k, o]
      v = W2[((size_t)hh * INFEAT + k) * 8 + o];
    }
  }
  B0s[idx] = (__bf16)v;
}

// pack [Wl1 | Wr1 | Wres1] -> bf16 swizzled fragment panel
__global__ void pack_w1_kernel(const float* __restrict__ Wl1, const float* __restrict__ Wr1,
                               const float* __restrict__ Wres1, __bf16* __restrict__ B1s) {
  size_t idx = (size_t)blockIdx.x * blockDim.x + threadIdx.x;
  if (idx >= (size_t)KP1 * NB1) return;
  int e    = (int)(idx & 15);
  int lane = (int)((idx >> 4) & 31);
  size_t tile = idx >> 9;
  int ct = (int)(tile % NCT1), kt = (int)(tile / NCT1);
  int k = kt * 32 + (lane >> 4) * 16 + e;
  int c = ct * 16 + (lane & 15);
  float v;
  if (c < 32)      v = Wl1[(size_t)k * 32 + c];
  else if (c < 64) v = Wr1[(size_t)k * 32 + (c - 32)];
  else             v = Wres1[(size_t)k * 32 + (c - 64)];
  B1s[idx] = (__bf16)v;
}

// ---------------------------------------------------------------------------
// Graph attention pieces
// ---------------------------------------------------------------------------
__global__ void fill_f32_kernel(float* __restrict__ p, float v, size_t count) {
  size_t idx = (size_t)blockIdx.x * blockDim.x + threadIdx.x;
  if (idx < count) p[idx] = v;
}

__device__ inline void atomicMaxFloat(float* addr, float val) {
  if (val >= 0.0f) atomicMax((int*)addr, __float_as_int(val));
  else             atomicMin((unsigned int*)addr, __float_as_uint(val));
}

// inner GAT per-node scores from Y0 cols [3072, 3104)
__global__ void inner_scores_kernel(const float* __restrict__ Y0, const float* __restrict__ a2,
                                    float* __restrict__ es, float* __restrict__ ed) {
  int idx = blockIdx.x * blockDim.x + threadIdx.x;
  if (idx >= NNODES * HEADS) return;
  int node = idx >> 2, h = idx & 3;
  const float* z = Y0 + (size_t)node * NB0 + 3072 + h * 8;
  float s = 0.f, d = 0.f;
#pragma unroll
  for (int o = 0; o < 8; ++o) { s += z[o] * a2[h * 16 + o]; d += z[o] * a2[h * 16 + 8 + o]; }
  es[idx] = s; ed[idx] = d;
}

__global__ void edge_logits_inner_kernel(const int* __restrict__ src, const int* __restrict__ dst,
                                         const float* __restrict__ es, const float* __restrict__ ed,
                                         float* __restrict__ logit, float* __restrict__ mbuf) {
  int idx = blockIdx.x * blockDim.x + threadIdx.x;
  if (idx >= NEDGES * HEADS) return;
  int e = idx >> 2, h = idx & 3;
  float v = es[src[e] * 4 + h] + ed[dst[e] * 4 + h];
  v = (v > 0.f) ? v : 0.2f * v;
  logit[idx] = v;
  atomicMaxFloat(&mbuf[dst[e] * 4 + h], v);
}

// GATv2: e = a . leaky_relu(fs[src] + fd[dst])
__global__ void edge_logits_gatv2_kernel(const int* __restrict__ src, const int* __restrict__ dst,
                                         const float* __restrict__ Y, const float* __restrict__ attn,
                                         float* __restrict__ logit, float* __restrict__ mbuf,
                                         int ldc, int fs_off, int fd_off, int d) {
  int idx = blockIdx.x * blockDim.x + threadIdx.x;
  if (idx >= NEDGES * HEADS) return;
  int e = idx >> 2, h = idx & 3;
  const float* fs = Y + (size_t)src[e] * ldc + fs_off + h * d;
  const float* fd = Y + (size_t)dst[e] * ldc + fd_off + h * d;
  const float* a  = attn + h * d;
  float acc = 0.f;
  for (int i = 0; i < d; ++i) {
    float v = fs[i] + fd[i];
    v = (v > 0.f) ? v : 0.2f * v;
    acc += a[i] * v;
  }
  logit[idx] = acc;
  atomicMaxFloat(&mbuf[dst[e] * 4 + h], acc);
}

__global__ void edge_exp_kernel(const int* __restrict__ dst, float* __restrict__ logit,
                                const float* __restrict__ mbuf, float* __restrict__ denom) {
  int idx = blockIdx.x * blockDim.x + threadIdx.x;
  if (idx >= NEDGES * HEADS) return;
  int e = idx >> 2, h = idx & 3;
  float m = mbuf[dst[e] * 4 + h];
  if (!(m > -1e37f && m < 1e37f)) m = 0.f;    // guard empty segments
  float p = expf(logit[idx] - m);
  logit[idx] = p;
  atomicAdd(&denom[dst[e] * 4 + h], p);
}

// out[dst, out_off + c] += fs[src, fs_off + c] * alpha,  c in [0, 4*d)
__global__ void edge_aggregate_kernel(const int* __restrict__ src, const int* __restrict__ dst,
                                      const float* __restrict__ Yin, const float* __restrict__ p,
                                      const float* __restrict__ denom, float* __restrict__ out,
                                      int ld_in, int fs_off, int ld_out, int out_off, int d) {
  size_t idx = (size_t)blockIdx.x * blockDim.x + threadIdx.x;
  size_t total = (size_t)NEDGES * HEADS * d;
  if (idx >= total) return;
  int hd = HEADS * d;
  int c = (int)(idx % hd);
  int e = (int)(idx / hd);
  int h = c / d;
  float alpha = p[e * 4 + h] / fmaxf(denom[dst[e] * 4 + h], 1e-9f);
  float v = Yin[(size_t)src[e] * ld_in + fs_off + c] * alpha;
  atomicAdd(&out[(size_t)dst[e] * ld_out + out_off + c], v);
}

__global__ void add_bias_kernel(float* __restrict__ Y, const float* __restrict__ b,
                                int n, int ldc, int off, int w) {
  size_t idx = (size_t)blockIdx.x * blockDim.x + threadIdx.x;
  if (idx >= (size_t)n * w) return;
  int node = (int)(idx / w), c = (int)(idx % w);
  Y[(size_t)node * ldc + off + c] += b[c];
}

// relu(Y0 residual columns) -> bf16 matrix [MPAD, KP1] for the layer-1 GEMM
__global__ void relu_to_bf16_kernel(const float* __restrict__ Y, __bf16* __restrict__ H,
                                    int ldc, int off, int w) {
  size_t idx = (size_t)blockIdx.x * blockDim.x + threadIdx.x;
  if (idx >= (size_t)MPAD * w) return;
  int row = (int)(idx / w), c = (int)(idx % w);
  float v = Y[(size_t)row * ldc + off + c];
  H[idx] = (__bf16)fmaxf(v, 0.f);
}

__global__ void relu_inplace_kernel(float* __restrict__ Y, int n, int ldc, int off, int w) {
  size_t idx = (size_t)blockIdx.x * blockDim.x + threadIdx.x;
  if (idx >= (size_t)n * w) return;
  int node = (int)(idx / w), c = (int)(idx % w);
  float* p = &Y[(size_t)node * ldc + off + c];
  *p = fmaxf(*p, 0.f);
}

// out[n, j] = concat(h3[n,:32], relu1[n,:32]) @ Wlin + blin
__global__ void final_kernel(const float* __restrict__ h3, const float* __restrict__ Y1,
                             const float* __restrict__ Wlin, const float* __restrict__ blin,
                             float* __restrict__ out) {
  int idx = blockIdx.x * blockDim.x + threadIdx.x;
  if (idx >= NNODES * OUTC) return;
  int node = idx / OUTC, j = idx % OUTC;
  float acc = blin[j];
  const float* a = h3 + (size_t)node * 32;
  const float* b = Y1 + (size_t)node * NB1 + 64;
#pragma unroll 8
  for (int c = 0; c < 32; ++c) acc += a[c] * Wlin[c * OUTC + j];
#pragma unroll 8
  for (int c = 0; c < 32; ++c) acc += b[c] * Wlin[(32 + c) * OUTC + j];
  out[idx] = acc;
}

// ---------------------------------------------------------------------------
// Host launcher
// ---------------------------------------------------------------------------
extern "C" void kernel_launch(void* const* d_in, const int* in_sizes, int n_in,
                              void* d_out, int out_size, void* d_ws, size_t ws_size,
                              hipStream_t stream) {
  (void)in_sizes; (void)n_in; (void)out_size; (void)ws_size;
  const float* feats = (const float*)d_in[0];
  const int*   src   = (const int*)d_in[1];
  const int*   dst   = (const int*)d_in[2];
  const float* tmask = (const float*)d_in[3];
  const float* amask = (const float*)d_in[4];
  const float* vmask = (const float*)d_in[5];
  const float* W2    = (const float*)d_in[6];
  const float* a2    = (const float*)d_in[7];
  const float* Wl0   = (const float*)d_in[8];
  const float* Wr0   = (const float*)d_in[9];
  const float* a0    = (const float*)d_in[10];
  const float* Wres0 = (const float*)d_in[11];
  const float* b0    = (const float*)d_in[12];
  const float* Wl1   = (const float*)d_in[13];
  const float* Wr1   = (const float*)d_in[14];
  const float* a1    = (const float*)d_in[15];
  const float* Wres1 = (const float*)d_in[16];
  const float* b1    = (const float*)d_in[17];
  const float* Wlin  = (const float*)d_in[18];
  const float* blin  = (const float*)d_in[19];
  float* out = (float*)d_out;

  // workspace carve-up (256B aligned)
  char* ws = (char*)d_ws;
  size_t off = 0;
  auto carve = [&](size_t bytes) -> char* {
    char* p = ws + off;
    off = (off + bytes + 255) & ~(size_t)255;
    return p;
  };
  __bf16* Xbf  = (__bf16*)carve((size_t)MPAD * KP0 * 2);
  __bf16* B0s  = (__bf16*)carve((size_t)KP0 * NB0 * 2);
  float*  Y0   = (float*)carve((size_t)MPAD * NB0 * 4);
  __bf16* H0bf = (__bf16*)carve((size_t)MPAD * KP1 * 2);
  __bf16* B1s  = (__bf16*)carve((size_t)KP1 * NB1 * 2);
  float*  Y1   = (float*)carve((size_t)MPAD * NB1 * 4);
  float*  mean = (float*)carve((size_t)INFEAT * 4);
  float*  es   = (float*)carve((size_t)NNODES * HEADS * 4);
  float*  ed   = (float*)carve((size_t)NNODES * HEADS * 4);
  float*  mbuf = (float*)carve((size_t)NNODES * HEADS * 4);
  float*  dnm  = (float*)carve((size_t)NNODES * HEADS * 4);
  float*  logit= (float*)carve((size_t)NEDGES * HEADS * 4);
  float*  h3   = (float*)carve((size_t)NNODES * 32 * 4);

  const int T = 256;
  const float NEG_INF = -__builtin_huge_valf();

  // ---- preprocessing + bf16 staging -------------------------------------
  col_mean_kernel<<<cdiv(INFEAT, T), T, 0, stream>>>(feats, mean);
  preprocess_kernel<<<MPAD, T, 0, stream>>>(feats, mean, tmask, amask, vmask, Xbf);
  pack_w0_kernel<<<cdiv((long long)KP0 * NB0, T), T, 0, stream>>>(Wl0, Wr0, Wres0, W2, B0s);
  pack_w1_kernel<<<cdiv((long long)KP1 * NB1, T), T, 0, stream>>>(Wl1, Wr1, Wres1, B1s);

  // ---- fused layer-0 + inner-GAT projections (WMMA) ---------------------
  {
    int Mt = MPAD / 64, Nt = NB0 / 32;                  // 157 x 97
    int blocks = cdiv((long long)Mt * Nt, 8);           // 8 waves / block
    wmma_gemm_kernel<<<blocks, T, 0, stream>>>(Xbf, B0s, Y0, Mt, Nt, KP0, KP0, NCT0, NB0);
  }

  // ---- inner GAT (classic) ----------------------------------------------
  inner_scores_kernel<<<cdiv(NNODES * HEADS, T), T, 0, stream>>>(Y0, a2, es, ed);
  fill_f32_kernel<<<cdiv(NNODES * HEADS, T), T, 0, stream>>>(mbuf, NEG_INF, (size_t)NNODES * HEADS);
  fill_f32_kernel<<<cdiv(NNODES * HEADS, T), T, 0, stream>>>(dnm, 0.f, (size_t)NNODES * HEADS);
  fill_f32_kernel<<<cdiv(NNODES * 32, T), T, 0, stream>>>(h3, 0.f, (size_t)NNODES * 32);
  edge_logits_inner_kernel<<<cdiv(NEDGES * HEADS, T), T, 0, stream>>>(src, dst, es, ed, logit, mbuf);
  edge_exp_kernel<<<cdiv(NEDGES * HEADS, T), T, 0, stream>>>(dst, logit, mbuf, dnm);
  edge_aggregate_kernel<<<cdiv((long long)NEDGES * HEADS * DIM1, T), T, 0, stream>>>(
      src, dst, Y0, logit, dnm, h3, NB0, 3072, 32, 0, DIM1);

  // ---- GATv2 layer 0 (aggregate into Y0 residual columns 2048..3071) ----
  add_bias_kernel<<<cdiv((long long)NNODES * 1024, T), T, 0, stream>>>(Y0, b0, NNODES, NB0, 2048, 1024);
  fill_f32_kernel<<<cdiv(NNODES * HEADS, T), T, 0, stream>>>(mbuf, NEG_INF, (size_t)NNODES * HEADS);
  fill_f32_kernel<<<cdiv(NNODES * HEADS, T), T, 0, stream>>>(dnm, 0.f, (size_t)NNODES * HEADS);
  edge_logits_gatv2_kernel<<<cdiv(NEDGES * HEADS, T), T, 0, stream>>>(
      src, dst, Y0, a0, logit, mbuf, NB0, 0, 1024, DIM0);
  edge_exp_kernel<<<cdiv(NEDGES * HEADS, T), T, 0, stream>>>(dst, logit, mbuf, dnm);
  edge_aggregate_kernel<<<cdiv((long long)NEDGES * HEADS * DIM0, T), T, 0, stream>>>(
      src, dst, Y0, logit, dnm, Y0, NB0, 0, NB0, 2048, DIM0);
  relu_to_bf16_kernel<<<cdiv((long long)MPAD * KP1, T), T, 0, stream>>>(Y0, H0bf, NB0, 2048, KP1);

  // ---- layer-1 projections (WMMA) ---------------------------------------
  {
    int Mt = MPAD / 64, Nt = NB1 / 32;                  // 157 x 3
    int blocks = cdiv((long long)Mt * Nt, 8);
    wmma_gemm_kernel<<<blocks, T, 0, stream>>>(H0bf, B1s, Y1, Mt, Nt, KP1, KP1, NCT1, NB1);
  }

  // ---- GATv2 layer 1 (aggregate into Y1 residual columns 64..95) --------
  add_bias_kernel<<<cdiv((long long)NNODES * 32, T), T, 0, stream>>>(Y1, b1, NNODES, NB1, 64, 32);
  fill_f32_kernel<<<cdiv(NNODES * HEADS, T), T, 0, stream>>>(mbuf, NEG_INF, (size_t)NNODES * HEADS);
  fill_f32_kernel<<<cdiv(NNODES * HEADS, T), T, 0, stream>>>(dnm, 0.f, (size_t)NNODES * HEADS);
  edge_logits_gatv2_kernel<<<cdiv(NEDGES * HEADS, T), T, 0, stream>>>(
      src, dst, Y1, a1, logit, mbuf, NB1, 0, 32, DIM1);
  edge_exp_kernel<<<cdiv(NEDGES * HEADS, T), T, 0, stream>>>(dst, logit, mbuf, dnm);
  edge_aggregate_kernel<<<cdiv((long long)NEDGES * HEADS * DIM1, T), T, 0, stream>>>(
      src, dst, Y1, logit, dnm, Y1, NB1, 0, NB1, 64, DIM1);
  relu_inplace_kernel<<<cdiv((long long)NNODES * 32, T), T, 0, stream>>>(Y1, NNODES, NB1, 64, 32);

  // ---- output projection --------------------------------------------------
  final_kernel<<<cdiv(NNODES * OUTC, T), T, 0, stream>>>(h3, Y1, Wlin, blin, out);
}
